// HebbianLearner_23871428232070
// MI455X (gfx1250) — matrix operand order repaired and checked by reading, not compiled, not run on hardware
//
#include <hip/hip_runtime.h>

// Sizes fixed by the reference
#define BATCH 32768
#define INSZ  1024
#define HID   1024
#define LR_SCALE (0.01f / 32768.0f)   // LEARNING_RATE * TEMPORAL_WEIGHT / BATCH

typedef __attribute__((ext_vector_type(16))) __bf16 v16bf;
typedef __attribute__((ext_vector_type(8)))  float  v8f;
typedef int v4i __attribute__((vector_size(16)));   // matches async builtin param

#define AS1 __attribute__((address_space(1)))
#define AS3 __attribute__((address_space(3)))

#if defined(__has_builtin)
#if __has_builtin(__builtin_amdgcn_global_load_async_to_lds_b128) && \
    __has_builtin(__builtin_amdgcn_s_wait_asynccnt)
#define USE_ASYNC_LDS 1
#endif
#if __has_builtin(__builtin_amdgcn_tanhf)
#define TANH(x) __builtin_amdgcn_tanhf(x)     // hardware V_TANH_F32
#endif
#endif
#ifndef USE_ASYNC_LDS
#define USE_ASYNC_LDS 0
#endif
#ifndef TANH
#define TANH(x) tanhf(x)
#endif

__device__ __forceinline__ unsigned short f2bf(float f) {
  unsigned int u = __float_as_uint(f);
  u += 0x7FFFu + ((u >> 16) & 1u);      // round-to-nearest-even
  return (unsigned short)(u >> 16);
}

// ---------------------------------------------------------------------------
// LDS tiles are stored so each WMMA lane reads CONTIGUOUS 16B chunks
// (ds_load_b128); row stride LDT=40 ushorts (80 B: 16B-aligned rows,
// 20-dword stride -> 16 distinct banks over a half-wave).
//
// A 16x32 bf16 (ISA 7.12.2): lane&15 = row M; lane>=16 -> K chunks at +8.
// B 32x16 bf16: lane&15 = col N; lane>=16 -> K chunks at +8 (tile stored [N][K]).
// ---------------------------------------------------------------------------
#define LDT 40

union Frag16 { uint4 q[2]; v16bf v; };

__device__ __forceinline__ v16bf load_a_frag(const unsigned short* lds, int lane) {
  const unsigned short* p = lds + (lane & 15) * LDT + ((lane >> 4) << 3);
  Frag16 f;
  f.q[0] = *reinterpret_cast<const uint4*>(p);        // K = klo .. klo+7
  f.q[1] = *reinterpret_cast<const uint4*>(p + 16);   // K = 16+klo .. 23+klo
  return f.v;
}

__device__ __forceinline__ v16bf load_b_frag(const unsigned short* lds, int lane) {
  const unsigned short* p = lds + (lane & 15) * LDT + ((lane >> 4) << 3);
  Frag16 f;
  f.q[0] = *reinterpret_cast<const uint4*>(p);        // K = klo .. klo+7
  f.q[1] = *reinterpret_cast<const uint4*>(p + 16);   // K = 16+klo .. 23+klo
  return f.v;
}

#if USE_ASYNC_LDS
__device__ __forceinline__ void async_copy16(const unsigned short* gp, void* lds_p) {
  __builtin_amdgcn_global_load_async_to_lds_b128(
      (AS1 v4i*)(uintptr_t)gp,
      (AS3 v4i*)(unsigned)(uintptr_t)lds_p,
      0, 0);
}
#endif

// Copy a 32(K) x 128(C) bf16 tile from global, transposed into LDS [C][K].
// 512 uint4 chunks, 2 per thread.
__device__ __forceinline__ void store_transposed_tile(
    unsigned short (*dst)[LDT], const unsigned short* __restrict__ g,
    int ldg, int tid) {
#pragma unroll
  for (int it = 0; it < 2; ++it) {
    const int c   = tid + it * 256;
    const int row = c >> 4;            // k-local 0..31
    const int cb  = (c & 15) * 8;      // col base 0..120
    uint4 d = *reinterpret_cast<const uint4*>(g + (size_t)row * ldg + cb);
    unsigned short t[8];
    *reinterpret_cast<uint4*>(t) = d;
#pragma unroll
    for (int j = 0; j < 8; ++j) dst[cb + j][row] = t[j];
  }
}

// ---- cast f32 -> bf16 (vectorized x4) -------------------------------------
__global__ void cast_bf16_kernel(const float* __restrict__ src,
                                 unsigned short* __restrict__ dst, int n4) {
  int i = blockIdx.x * blockDim.x + threadIdx.x;
  if (i < n4) {
    float4 f = reinterpret_cast<const float4*>(src)[i];
    ushort4 o;
    o.x = f2bf(f.x); o.y = f2bf(f.y); o.z = f2bf(f.z); o.w = f2bf(f.w);
    reinterpret_cast<ushort4*>(dst)[i] = o;
  }
}

// ---- GEMM1: hidden = tanh(x @ W), also emit hidden as bf16 ----------------
// Block tile 128(M) x 128(N), K step 32. 8 waves as 4(M) x 2(N);
// wave tile 32x64 = 2 A-frags x 4 B-frags -> 8 WMMAs per 12 ds_load_b128.
#define G1_MT 128
#define G1_NT 128
#define G1_KT 32

__global__ __launch_bounds__(256)
void gemm1_tanh_kernel(const unsigned short* __restrict__ xb,
                       const unsigned short* __restrict__ wb,
                       float* __restrict__ hidden,
                       unsigned short* __restrict__ hbf) {
  __shared__ unsigned short Xs[2][G1_MT][LDT];   // [M][K] row-major (A layout)
  __shared__ unsigned short Wt[2][G1_NT][LDT];   // [N][K] transposed (B layout)

  const int tid  = threadIdx.x;
  const int lane = tid & 31;
  const int wave = tid >> 5;
  const int m0   = blockIdx.y * G1_MT;
  const int n0   = blockIdx.x * G1_NT;
  const int wM   = (wave >> 1) * 32;   // 4 waves along M
  const int wN   = (wave & 1) * 64;    // 2 waves along N

  v8f acc[2][4];
#pragma unroll
  for (int a = 0; a < 2; ++a)
#pragma unroll
    for (int b = 0; b < 4; ++b)
#pragma unroll
      for (int r = 0; r < 8; ++r) acc[a][b][r] = 0.0f;

  auto loadX = [&](int buf, int kt) {
    const int k0 = kt * G1_KT;
#if USE_ASYNC_LDS
    // 128 rows x 64B = 512 x 16B chunks, DMA'd straight into LDS (ASYNCcnt).
#pragma unroll
    for (int it = 0; it < 2; ++it) {
      const int c   = tid + it * 256;
      const int row = c >> 2;
      const int q   = (c & 3) * 8;                 // ushort offset in row
      async_copy16(xb + (size_t)(m0 + row) * INSZ + k0 + q, &Xs[buf][row][q]);
    }
#else
    const int row = tid >> 1;
    const int ch  = (tid & 1) * 16;
    const unsigned short* gp = xb + (size_t)(m0 + row) * INSZ + k0 + ch;
    uint4 d0 = *reinterpret_cast<const uint4*>(gp);
    uint4 d1 = *reinterpret_cast<const uint4*>(gp + 8);
    *reinterpret_cast<uint4*>(&Xs[buf][row][ch])     = d0;
    *reinterpret_cast<uint4*>(&Xs[buf][row][ch + 8]) = d1;
#endif
  };
  auto loadW = [&](int buf, int kt) {  // transpose [K][N] -> [N][K] on store
    store_transposed_tile(Wt[buf], wb + (size_t)(kt * G1_KT) * HID + n0, HID, tid);
  };

  const int NK = INSZ / G1_KT;          // 32
  loadX(0, 0); loadW(0, 0);
#if USE_ASYNC_LDS
  __builtin_amdgcn_s_wait_asynccnt(0);
#endif
  __syncthreads();
  int buf = 0;
  for (int kt = 0; kt < NK; ++kt) {
    if (kt + 1 < NK) { loadX(buf ^ 1, kt + 1); loadW(buf ^ 1, kt + 1); }
    v16bf af[2], bfr[4];
#pragma unroll
    for (int mi = 0; mi < 2; ++mi)
      af[mi] = load_a_frag(&Xs[buf][wM + mi * 16][0], lane);
#pragma unroll
    for (int ni = 0; ni < 4; ++ni)
      bfr[ni] = load_b_frag(&Wt[buf][wN + ni * 16][0], lane);
#pragma unroll
    for (int mi = 0; mi < 2; ++mi)
#pragma unroll
      for (int ni = 0; ni < 4; ++ni)
        acc[mi][ni] = __builtin_amdgcn_wmma_f32_16x16x32_bf16(
            false, af[mi], false, bfr[ni], (short)0, acc[mi][ni], false, false);
#if USE_ASYNC_LDS
    __builtin_amdgcn_s_wait_asynccnt(0);
#endif
    __syncthreads();
    buf ^= 1;
  }

  // C layout: lane&15 = N, lanes>=16 -> M += 8; VGPR r -> M = r
  const int n  = lane & 15;
  const int mb = (lane >> 4) << 3;
#pragma unroll
  for (int mi = 0; mi < 2; ++mi)
#pragma unroll
    for (int ni = 0; ni < 4; ++ni)
#pragma unroll
      for (int r = 0; r < 8; ++r) {
        const int row = m0 + wM + mi * 16 + mb + r;
        const int col = n0 + wN + ni * 16 + n;
        const float h = TANH(acc[mi][ni][r]);
        const size_t idx = (size_t)row * HID + col;
        hidden[idx] = h;
        hbf[idx]    = f2bf(h);
      }
}

// ---- GEMM2: delta_part = x^T @ hidden over one K chunk --------------------
// Block tile 128(I) x 128(H), K step 32, split-K = 8 (deterministic partials).
#define G2_T 128
#define G2_KT 32
#define NSPLIT 8
#define KCHUNK (BATCH / NSPLIT)   // 4096

__global__ __launch_bounds__(256)
void gemm2_kernel(const unsigned short* __restrict__ xb,
                  const unsigned short* __restrict__ hbf,
                  float* __restrict__ dpart) {
  __shared__ unsigned short Xt[2][G2_T][LDT];   // [I][K] : A = x^T (row per lane)
  __shared__ unsigned short Ht[2][G2_T][LDT];   // [H][K] : B layout [N][K]

  const int tid  = threadIdx.x;
  const int lane = tid & 31;
  const int wave = tid >> 5;
  const int h0   = blockIdx.x * G2_T;
  const int i0   = blockIdx.y * G2_T;
  const int kb0  = blockIdx.z * KCHUNK;
  const int wM   = (wave >> 1) * 32;   // I offset, 4 waves
  const int wN   = (wave & 1) * 64;    // H offset, 2 waves

  v8f acc[2][4];
#pragma unroll
  for (int a = 0; a < 2; ++a)
#pragma unroll
    for (int b = 0; b < 4; ++b)
#pragma unroll
      for (int r = 0; r < 8; ++r) acc[a][b][r] = 0.0f;

  auto loadTiles = [&](int buf, int kt) {  // both tiles transposed on store
    const size_t k0 = kb0 + kt * G2_KT;
    store_transposed_tile(Xt[buf], xb  + k0 * INSZ + i0, INSZ, tid);
    store_transposed_tile(Ht[buf], hbf + k0 * HID  + h0, HID,  tid);
  };

  const int NK = KCHUNK / G2_KT;       // 128
  loadTiles(0, 0);
  __syncthreads();
  int buf = 0;
  for (int kt = 0; kt < NK; ++kt) {
    if (kt + 1 < NK) loadTiles(buf ^ 1, kt + 1);
    v16bf af[2], bfr[4];
#pragma unroll
    for (int mi = 0; mi < 2; ++mi)
      af[mi] = load_a_frag(&Xt[buf][wM + mi * 16][0], lane);
#pragma unroll
    for (int ni = 0; ni < 4; ++ni)
      bfr[ni] = load_b_frag(&Ht[buf][wN + ni * 16][0], lane);
#pragma unroll
    for (int mi = 0; mi < 2; ++mi)
#pragma unroll
      for (int ni = 0; ni < 4; ++ni)
        acc[mi][ni] = __builtin_amdgcn_wmma_f32_16x16x32_bf16(
            false, af[mi], false, bfr[ni], (short)0, acc[mi][ni], false, false);
    __syncthreads();
    buf ^= 1;
  }

  float* out = dpart + (size_t)blockIdx.z * INSZ * HID;
  const int n  = lane & 15;
  const int mb = (lane >> 4) << 3;
#pragma unroll
  for (int mi = 0; mi < 2; ++mi)
#pragma unroll
    for (int ni = 0; ni < 4; ++ni)
#pragma unroll
      for (int r = 0; r < 8; ++r) {
        const int row = i0 + wM + mi * 16 + mb + r;
        const int col = h0 + wN + ni * 16 + n;
        out[(size_t)row * HID + col] = acc[mi][ni][r];
      }
}

// ---- finalize: new_W = clip(W + lr/B * sum(partials), -1, 1) --------------
__global__ void finalize_w_kernel(const float* __restrict__ W,
                                  const float* __restrict__ dpart,
                                  float* __restrict__ outW) {
  int i = blockIdx.x * blockDim.x + threadIdx.x;
  if (i < INSZ * HID) {
    const size_t sz = (size_t)INSZ * HID;
    float s = 0.0f;
#pragma unroll
    for (int p = 0; p < NSPLIT; ++p) s += dpart[i + (size_t)p * sz];
    float w = W[i] + LR_SCALE * s;
    w = fminf(1.0f, fmaxf(-1.0f, w));
    outW[i] = w;
  }
}

extern "C" void kernel_launch(void* const* d_in, const int* in_sizes, int n_in,
                              void* d_out, int out_size, void* d_ws, size_t ws_size,
                              hipStream_t stream) {
  (void)in_sizes; (void)n_in; (void)out_size; (void)ws_size;
  const float* x = (const float*)d_in[0];   // [32768,1024] f32
  const float* W = (const float*)d_in[1];   // [1024,1024]  f32

  float* hidden = (float*)d_out;                          // [32768,1024]
  float* newW   = (float*)d_out + (size_t)BATCH * HID;    // [1024,1024]

  // Workspace layout (~162 MB): x_bf16 | W_bf16 | hidden_bf16 | delta partials
  unsigned short* xb = (unsigned short*)d_ws;
  unsigned short* wb = xb + (size_t)BATCH * INSZ;
  unsigned short* hb = wb + (size_t)INSZ * HID;
  float*       dpart = (float*)(hb + (size_t)BATCH * HID);

  cast_bf16_kernel<<<(BATCH * INSZ / 4 + 255) / 256, 256, 0, stream>>>(x, xb, BATCH * INSZ / 4);
  cast_bf16_kernel<<<(INSZ * HID / 4 + 255) / 256, 256, 0, stream>>>(W, wb, INSZ * HID / 4);

  dim3 g1(HID / G1_NT, BATCH / G1_MT);          // 8 x 256 blocks
  gemm1_tanh_kernel<<<g1, 256, 0, stream>>>(xb, wb, hidden, hb);

  dim3 g2(HID / G2_T, INSZ / G2_T, NSPLIT);     // 8 x 8 x 8 blocks
  gemm2_kernel<<<g2, 256, 0, stream>>>(xb, hb, dpart);

  finalize_w_kernel<<<(INSZ * HID + 255) / 256, 256, 0, stream>>>(W, dpart, newW);
}
